// MultiHeadSelfAttention_6236292514379
// MI455X (gfx1250) — matrix operand order repaired and checked by reading, not compiled, not run on hardware
//
#include <hip/hip_runtime.h>

// ---------------------------------------------------------------------------
// MI455X (gfx1250) self-attention, bf16 WMMA + Tensor Data Mover pipeline.
// ~275 GFLOP vs ~0.5 GB HBM traffic -> compute bound -> every GEMM uses
// v_wmma_f32_16x16x32_bf16. All GEMMs canonicalized to C = A[M,K] @ Bt[N,K]^T
// so both LDS tiles are plain 2-D row blocks, loaded by TENSOR_LOAD_TO_LDS
// with TDM padding (16B pad per 64B row => padded LDS stride of 40 elems).
// ---------------------------------------------------------------------------

typedef __attribute__((ext_vector_type(16))) __bf16 v16bf;
typedef __attribute__((ext_vector_type(8)))  float  v8f;
typedef __attribute__((ext_vector_type(4)))  unsigned int v4u;
typedef __attribute__((ext_vector_type(8)))  int v8i_t;
typedef __attribute__((ext_vector_type(4)))  int v4i_t;

#if __has_builtin(__builtin_amdgcn_tensor_load_to_lds)
#define HAVE_TDM 1
#else
#define HAVE_TDM 0
#endif

static __device__ __forceinline__ unsigned short f32_to_bf16(float f) {
    unsigned int u = __float_as_uint(f);
    u += 0x7fffu + ((u >> 16) & 1u);      // branchless round-to-nearest-even
    return (unsigned short)(u >> 16);
}

__global__ void cvt_f32_bf16(const float* __restrict__ in,
                             unsigned short* __restrict__ out, long n) {
    long i = (long)blockIdx.x * blockDim.x + threadIdx.x;
    long stride = (long)gridDim.x * blockDim.x;
    for (; i < n; i += stride) out[i] = f32_to_bf16(in[i]);
}

// out[c][r] = bf16(in[r][c]); 32x32 tiles, block (32,8)
__global__ __launch_bounds__(256) void cvt_transpose_bf16(
    const float* __restrict__ in, unsigned short* __restrict__ out,
    int rows, int cols)
{
    __shared__ float t[32][33];
    int tx = threadIdx.x, ty = threadIdx.y;
    int r0 = blockIdx.y * 32, c0 = blockIdx.x * 32;
#pragma unroll
    for (int j = 0; j < 32; j += 8)
        t[ty + j][tx] = in[(long)(r0 + ty + j) * cols + (c0 + tx)];
    __syncthreads();
#pragma unroll
    for (int j = 0; j < 32; j += 8)
        out[(long)(c0 + ty + j) * rows + (r0 + tx)] = f32_to_bf16(t[tx][ty + j]);
}

#define BM 128
#define BN 128
#define BK 32
#define LDT 40   // padded LDS row stride (elements) = 32 data + 8 pad (TDM pad)

union V16 { uint4 q[2]; v16bf v; };

#if HAVE_TDM
// Issue one TDM 2-D tile load: BM rows x BK bf16 elems from a row-major
// matrix (row stride = stride_elems) into LDS at lds_byte_off, with the TDM
// inserting 4 DWORDs (8 elems) of pad after every 16 DWORDs (32 elems) so the
// LDS row stride becomes LDT = 40 elements.
static __device__ __forceinline__ void tdm_load_tile(
    const unsigned short* gptr, unsigned lds_byte_off,
    int stride_elems, int rows_bound)
{
    unsigned long long ga = (unsigned long long)(size_t)gptr;
    v4u g0;
    g0[0] = 1u;                                            // count=1, user D#
    g0[1] = lds_byte_off;                                  // LDS byte address
    g0[2] = (unsigned)ga;                                  // global addr lo
    g0[3] = (unsigned)((ga >> 32) & 0x1FFFFFFu) | (2u << 30); // addr hi | type=2

    unsigned td0 = (unsigned)stride_elems;                 // tensor_dim0
    unsigned td1 = (unsigned)rows_bound;                   // tensor_dim1
    unsigned st0 = (unsigned)stride_elems;                 // tensor_dim0_stride
    v8i_t g1;
    g1[0] = (int)((1u << 16)        // data_size = 1 -> 2-byte elements
                | (1u << 20)        // pad_enable
                | (3u << 22)        // pad_interval: 16 DWORDs (= 32 elems)
                | (3u << 25));      // pad_amount:    4 DWORDs (=  8 elems)
    g1[1] = (int)((td0 & 0xFFFFu) << 16);
    g1[2] = (int)((td0 >> 16) | ((td1 & 0xFFFFu) << 16));
    g1[3] = (int)((td1 >> 16) | ((unsigned)BK << 16));     // tile_dim0 = 32
    g1[4] = (int)(unsigned)BM;                             // tile_dim1 = 128
    g1[5] = (int)st0;                                      // stride[31:0]
    g1[6] = 0;                                             // stride[47:32]
    g1[7] = 0;
    v4i_t g2 = {};
    v4i_t g3 = {};
#if defined(__clang_major__) && __clang_major__ >= 23
    v8i_t gx = {};
    __builtin_amdgcn_tensor_load_to_lds(g0, g1, g2, g3, gx, 0);
#else
    __builtin_amdgcn_tensor_load_to_lds(g0, g1, g2, g3, 0);
#endif
}
#endif

// Fragment loads per ISA 16-bit layouts + 8 WMMAs on one staged tile pair.
static __device__ __forceinline__ void frag_mma(
    const unsigned short* __restrict__ Asb, const unsigned short* __restrict__ Bsb,
    v8f acc[4][2], int wm, int wn, int l15, int lhalf)
{
    V16 af[4], bfr[2];
#pragma unroll
    for (int mt = 0; mt < 4; mt++) {
        const unsigned short* pA = &Asb[(wm * 64 + mt * 16 + l15) * LDT + lhalf * 8];
        af[mt].q[0] = *(const uint4*)pA;         // K = lhalf*8  + [0..7]
        af[mt].q[1] = *(const uint4*)(pA + 16);  // K = 16+lhalf*8 + [0..7]
    }
#pragma unroll
    for (int nt = 0; nt < 2; nt++) {
        const unsigned short* pB = &Bsb[(wn * 32 + nt * 16 + l15) * LDT + lhalf * 8];
        bfr[nt].q[0] = *(const uint4*)pB;        // K = lhalf*8  + [0..7]
        bfr[nt].q[1] = *(const uint4*)(pB + 16); // K = 16+lhalf*8 + [0..7]
    }
#pragma unroll
    for (int mt = 0; mt < 4; mt++)
#pragma unroll
        for (int nt = 0; nt < 2; nt++)
            acc[mt][nt] = __builtin_amdgcn_wmma_f32_16x16x32_bf16(
                false, af[mt].v, false, bfr[nt].v,
                (short)0, acc[mt][nt], false, false);
}

// C[M,N] = alpha * A[M,K] @ Bt[N,K]^T + bias   (A, Bt bf16 row-major)
// TRANS_OUT: store C transposed as [N,M] (row stride ldc).
template <bool TRANS_OUT, bool BF16_OUT>
__global__ __launch_bounds__(256) void wmma_gemm_tn(
    const unsigned short* __restrict__ A, const unsigned short* __restrict__ Bt,
    const float* __restrict__ bias, void* __restrict__ Cout,
    int M, int N, int K, int lda, int ldb, int ldc,
    long sA, long sB, long sC, float alpha)
{
    __shared__ __align__(16) unsigned short As[2][BM * LDT];
    __shared__ __align__(16) unsigned short Bs[2][BN * LDT];

    const int tid   = threadIdx.x;
    const int lane  = tid & 31;
    const int wave  = tid >> 5;
    const int wm    = wave >> 2;   // 0..1
    const int wn    = wave & 3;    // 0..3
    const int lhalf = lane >> 4;
    const int l15   = lane & 15;

    const int m0 = blockIdx.y * BM;
    const int n0 = blockIdx.x * BN;
    const int bz = blockIdx.z;

    const unsigned short* Ab = A  + (long)bz * sA + (long)m0 * lda;
    const unsigned short* Bb = Bt + (long)bz * sB + (long)n0 * ldb;

    v8f zero = {};
    v8f acc[4][2];
#pragma unroll
    for (int i = 0; i < 4; i++) { acc[i][0] = zero; acc[i][1] = zero; }

    const int nk = K / BK;

#if HAVE_TDM
    const unsigned ldsA0 = (unsigned)(size_t)&As[0][0];
    const unsigned ldsB0 = (unsigned)(size_t)&Bs[0][0];
    const unsigned tilebytes = BM * LDT * 2;

    if (wave == 0) {                       // one wave drives the DMA engine
        tdm_load_tile(Ab, ldsA0, lda, M - m0);
        tdm_load_tile(Bb, ldsB0, ldb, N - n0);
        __builtin_amdgcn_s_wait_tensorcnt(0);
    }
    __syncthreads();

    for (int kt = 0; kt < nk; kt++) {
        const int cur = kt & 1, nxt = cur ^ 1;
        if (wave == 0 && (kt + 1) < nk) {  // async prefetch of next tile pair
            tdm_load_tile(Ab + (kt + 1) * BK, ldsA0 + nxt * tilebytes, lda, M - m0);
            tdm_load_tile(Bb + (kt + 1) * BK, ldsB0 + nxt * tilebytes, ldb, N - n0);
        }
        frag_mma(&As[cur][0], &Bs[cur][0], acc, wm, wn, l15, lhalf);
        if (wave == 0) __builtin_amdgcn_s_wait_tensorcnt(0);
        __syncthreads();
    }
#else
    // Fallback: cooperative vector loads (2 x uint4 per thread per tile).
    const int rT = tid >> 1;
    const int hT = tid & 1;
    const unsigned short* gA = Ab + (long)rT * lda + hT * 16;
    const unsigned short* gB = Bb + (long)rT * ldb + hT * 16;
    unsigned short* sAw = &As[0][rT * LDT + hT * 16];
    unsigned short* sBw = &Bs[0][rT * LDT + hT * 16];
    for (int kt = 0; kt < nk; kt++) {
        const int k0 = kt * BK;
        uint4 a0 = *(const uint4*)(gA + k0);
        uint4 a1 = *(const uint4*)(gA + k0 + 8);
        uint4 b0 = *(const uint4*)(gB + k0);
        uint4 b1 = *(const uint4*)(gB + k0 + 8);
        ((uint4*)sAw)[0] = a0; ((uint4*)sAw)[1] = a1;
        ((uint4*)sBw)[0] = b0; ((uint4*)sBw)[1] = b1;
        __builtin_prefetch(gA + k0 + BK, 0, 0);
        __syncthreads();
        frag_mma(&As[0][0], &Bs[0][0], acc, wm, wn, l15, lhalf);
        __syncthreads();
    }
#endif

    // Epilogue: C/D fragment layout: VGPR r -> M = r + 8*lanehalf, N = lane&15
#pragma unroll
    for (int mt = 0; mt < 4; mt++)
#pragma unroll
        for (int nt = 0; nt < 2; nt++) {
            int n = n0 + wn * 32 + nt * 16 + l15;
            float bb = (bias != nullptr) ? bias[n] : 0.0f;
#pragma unroll
            for (int r = 0; r < 8; r++) {
                int m = m0 + wm * 64 + mt * 16 + 8 * lhalf + r;
                float v = acc[mt][nt][r] * alpha + bb;
                long off = TRANS_OUT ? ((long)bz * sC + (long)n * ldc + m)
                                     : ((long)bz * sC + (long)m * ldc + n);
                if (BF16_OUT) ((unsigned short*)Cout)[off] = f32_to_bf16(v);
                else          ((float*)Cout)[off]          = v;
            }
        }
}

// Row softmax over fp32 scores (row length 2048) -> bf16 probabilities.
__global__ __launch_bounds__(256) void softmax2048(const float* __restrict__ S,
                                                   unsigned short* __restrict__ P) {
    const int NN = 2048;
    long row = blockIdx.x;
    const float* s = S + row * (long)NN;
    unsigned short* p = P + row * (long)NN;
    int tid = threadIdx.x;
    __shared__ float red[8];

    float ev[8];
    float m = -3.4e38f;
#pragma unroll
    for (int j = 0; j < 8; j++) { ev[j] = s[tid + j * 256]; m = fmaxf(m, ev[j]); }
#pragma unroll
    for (int off = 16; off; off >>= 1) m = fmaxf(m, __shfl_xor(m, off, 32));
    if ((tid & 31) == 0) red[tid >> 5] = m;
    __syncthreads();
    float gm = red[0];
#pragma unroll
    for (int i = 1; i < 8; i++) gm = fmaxf(gm, red[i]);
    __syncthreads();

    float sum = 0.f;
#pragma unroll
    for (int j = 0; j < 8; j++) { ev[j] = __expf(ev[j] - gm); sum += ev[j]; }
#pragma unroll
    for (int off = 16; off; off >>= 1) sum += __shfl_xor(sum, off, 32);
    if ((tid & 31) == 0) red[tid >> 5] = sum;
    __syncthreads();
    float gs = 0.f;
#pragma unroll
    for (int i = 0; i < 8; i++) gs += red[i];
    float inv = 1.0f / gs;
#pragma unroll
    for (int j = 0; j < 8; j++) p[tid + j * 256] = f32_to_bf16(ev[j] * inv);
}

extern "C" void kernel_launch(void* const* d_in, const int* in_sizes, int n_in,
                              void* d_out, int out_size, void* d_ws, size_t ws_size,
                              hipStream_t stream)
{
    const int  Bb = 8, Nn = 2048, Dd = 1024;
    const long MD = (long)Bb * Nn;              // 16384 total rows

    const float* X  = (const float*)d_in[0];
    const float* Wq = (const float*)d_in[1];
    const float* bq = (const float*)d_in[2];
    const float* Wk = (const float*)d_in[3];
    const float* bk = (const float*)d_in[4];
    const float* Wv = (const float*)d_in[5];
    const float* bv = (const float*)d_in[6];
    const float* Wo = (const float*)d_in[7];
    const float* bo = (const float*)d_in[8];

    char* p = (char*)d_ws;
    auto alloc = [&](size_t bytes) {
        char* r = p; p += (bytes + 255) & ~(size_t)255; return r;
    };
    unsigned short* Xb  = (unsigned short*)alloc((size_t)MD * Dd * 2);
    unsigned short* Wqt = (unsigned short*)alloc((size_t)Dd * Dd * 2);  // W^T
    unsigned short* Wkt = (unsigned short*)alloc((size_t)Dd * Dd * 2);
    unsigned short* Wvt = (unsigned short*)alloc((size_t)Dd * Dd * 2);
    unsigned short* Wot = (unsigned short*)alloc((size_t)Dd * Dd * 2);
    unsigned short* Qb  = (unsigned short*)alloc((size_t)MD * Dd * 2);
    unsigned short* Kb  = (unsigned short*)alloc((size_t)MD * Dd * 2);
    unsigned short* Vt  = (unsigned short*)alloc((size_t)MD * Dd * 2);  // [b][d][n]
    float*          Sc  = (float*)         alloc((size_t)Bb * Nn * Nn * 4);
    unsigned short* Pp  = (unsigned short*)alloc((size_t)Bb * Nn * Nn * 2);
    unsigned short* Ob  = (unsigned short*)alloc((size_t)MD * Dd * 2);

    // 1) conversions: X plain, weights transposed to [out_dim][in_dim]
    cvt_f32_bf16<<<4096, 256, 0, stream>>>(X, Xb, MD * Dd);
    dim3 tb(32, 8), tg(Dd / 32, Dd / 32);
    cvt_transpose_bf16<<<tg, tb, 0, stream>>>(Wq, Wqt, Dd, Dd);
    cvt_transpose_bf16<<<tg, tb, 0, stream>>>(Wk, Wkt, Dd, Dd);
    cvt_transpose_bf16<<<tg, tb, 0, stream>>>(Wv, Wvt, Dd, Dd);
    cvt_transpose_bf16<<<tg, tb, 0, stream>>>(Wo, Wot, Dd, Dd);

    dim3 blk(256);

    // 2) Q/K projections: X @ Wt^T + bias -> bf16 [16384,1024]
    dim3 gP(Dd / BN, MD / BM, 1);
    wmma_gemm_tn<false, true><<<gP, blk, 0, stream>>>(Xb, Wqt, bq, Qb,
        (int)MD, Dd, Dd, Dd, Dd, Dd, 0, 0, 0, 1.0f);
    wmma_gemm_tn<false, true><<<gP, blk, 0, stream>>>(Xb, Wkt, bk, Kb,
        (int)MD, Dd, Dd, Dd, Dd, Dd, 0, 0, 0, 1.0f);
    // V projection, stored transposed per batch: Vt[b][d][token]
    dim3 gV(Dd / BN, Nn / BM, Bb);
    wmma_gemm_tn<true, true><<<gV, blk, 0, stream>>>(Xb, Wvt, bv, Vt,
        Nn, Dd, Dd, Dd, Dd, Nn,
        (long)Nn * Dd, 0, (long)Dd * Nn, 1.0f);

    // 3) scores = (Q @ K^T) / sqrt(D): batched, fp32
    dim3 gS(Nn / BN, Nn / BM, Bb);
    wmma_gemm_tn<false, false><<<gS, blk, 0, stream>>>(Qb, Kb, nullptr, Sc,
        Nn, Nn, Dd, Dd, Dd, Nn,
        (long)Nn * Dd, (long)Nn * Dd, (long)Nn * Nn, 0.03125f);

    // 4) softmax rows -> bf16 P
    softmax2048<<<Bb * Nn, 256, 0, stream>>>(Sc, Pp);

    // 5) O = P @ V = P @ Vt^T : batched [2048,2048] x [1024,2048]^T -> bf16
    dim3 gO(Dd / BN, Nn / BM, Bb);
    wmma_gemm_tn<false, true><<<gO, blk, 0, stream>>>(Pp, Vt, nullptr, Ob,
        Nn, Dd, Nn, Nn, Nn, Dd,
        (long)Nn * Nn, (long)Dd * Nn, (long)Nn * Dd, 1.0f);

    // 6) out = O @ Wo^T^T ... = O @ Wot^T + bo -> fp32 d_out
    dim3 gF(Dd / BN, MD / BM, 1);
    wmma_gemm_tn<false, false><<<gF, blk, 0, stream>>>(Ob, Wot, bo, d_out,
        (int)MD, Dd, Dd, Dd, Dd, Dd, 0, 0, 0, 1.0f);
}
